// UnifiedModel_3934190043556
// MI455X (gfx1250) — compile-verified, matrix operand order checked by reading, not run on hardware
//
#include <hip/hip_runtime.h>
#include <hip/hip_bf16.h>
#include <cstdint>
#include <cstddef>

// ---------------------------------------------------------------------------
// Model dims (from the reference)
// ---------------------------------------------------------------------------
#define N_DRUG   20000
#define N_RNA    40000
#define N_RNA_G  50000
#define BATCH    32768
#define D_DRUG   320
#define D_RNASIM 400
#define HID      256
#define OUTD     128
#define RNA_FEAT 256
#define DNN_HID  1280
#define PROJ     512
#define NE       400000
#define N_ATOMS  200000
#define E_MOL    400000
#define N_UNIQ   8000
#define ATOM_DIM 78

// ---------------------------------------------------------------------------
// bf16 helpers / WMMA types (gfx1250: wave32, v_wmma_f32_16x16x32_bf16)
// ---------------------------------------------------------------------------
typedef __attribute__((ext_vector_type(16))) __bf16 bf16x16;
typedef __attribute__((ext_vector_type(8)))  float  floatx8;

__device__ __forceinline__ unsigned short f32_to_bf16(float f) {
  unsigned u = __float_as_uint(f);
  u += 0x7FFFu + ((u >> 16) & 1u);   // round-to-nearest-even
  return (unsigned short)(u >> 16);
}

union Frag { bf16x16 v; uint2 q[4]; unsigned short u[16]; };

// ---------------------------------------------------------------------------
// Tiled GEMM: C[M,N] (+)= act( A[M,K](lda) @ W[K,N] + bias )
// Block = 256 threads = 8 waves; tile 128x64, K-step 32.
// Wave grid 4x2: each wave owns 32 rows x 32 cols -> 4 WMMA accumulators.
// A tile As[128][40] (row-major, bf16); W tile stored TRANSPOSED Bs[n][k]
// so both A-rows and B-columns are two contiguous 16B K-chunks per lane:
//   K(h,lane) = (h<8?0:16) + (lane>=16?8:0) + h%8   (ISA 7.12.2 bf16 layout)
// -> fragment loads are 4x ds_load_b64, no 16-bit LDS traffic.
// ---------------------------------------------------------------------------
__global__ __launch_bounds__(256)
void gemm_wmma_bf16(const float* __restrict__ A, int lda,
                    const float* __restrict__ W,          // [K,N], ldw = N
                    const float* __restrict__ bias,       // nullable
                    float* __restrict__ C, int ldc,
                    int M, int N, int K, int act, int accum) {
  __shared__ unsigned short As[128][40];   // 80B row stride: 8B aligned, good bank spread
  __shared__ unsigned short Bs[64][40];    // transposed: [n][k]

  const int tid  = threadIdx.x;
  const int lane = tid & 31;
  const int wid  = tid >> 5;
  const int wm   = wid & 3;      // 4 row groups of 32
  const int wn   = wid >> 2;     // 2 col groups of 32

  const int m0 = blockIdx.y * 128;
  const int n0 = blockIdx.x * 64;

  floatx8 c00 = {}, c01 = {}, c10 = {}, c11 = {};

  const int rm = lane & 15;
  const int kh = (lane >> 4) << 3;        // 0 or 8

  const bool aVec  = ((lda & 3) == 0);
  const bool mfull = (m0 + 128 <= M);
  const bool nfull = (n0 + 64 <= N) && ((N & 3) == 0);

  for (int k0 = 0; k0 < K; k0 += 32) {
    const bool kfull = (k0 + 32 <= K);

    // ---- stage A tile 128x32 (4 float4 pieces per thread) ----
    if (mfull && kfull && aVec) {
#pragma unroll
      for (int i = 0; i < 4; ++i) {
        int p = tid + (i << 8);                  // 0..1023
        int r = p >> 3, c = (p & 7) << 2;
        const float4 va = *(const float4*)(A + (size_t)(m0 + r) * lda + k0 + c);
        uint2 w;
        w.x = (unsigned)f32_to_bf16(va.x) | ((unsigned)f32_to_bf16(va.y) << 16);
        w.y = (unsigned)f32_to_bf16(va.z) | ((unsigned)f32_to_bf16(va.w) << 16);
        *(uint2*)&As[r][c] = w;
      }
    } else {
      // edge path: clamp addresses (always in-bounds loads), select zero after
#pragma unroll
      for (int i = 0; i < 4; ++i) {
        int p = tid + (i << 8);
        int r = p >> 3, c = (p & 7) << 2;
        int gm  = m0 + r;
        int gmc = gm < M ? gm : (M - 1);
        unsigned short h4[4];
#pragma unroll
        for (int j = 0; j < 4; ++j) {
          int gk  = k0 + c + j;
          int gkc = gk < K ? gk : (K - 1);
          float v = A[(size_t)gmc * lda + gkc];
          h4[j] = (gm < M && gk < K) ? f32_to_bf16(v) : (unsigned short)0;
        }
        uint2 w;
        w.x = h4[0] | ((unsigned)h4[1] << 16);
        w.y = h4[2] | ((unsigned)h4[3] << 16);
        *(uint2*)&As[r][c] = w;
      }
    }

    // ---- stage W tile 32x64 transposed into Bs[n][k] ----
    if (kfull && nfull) {
#pragma unroll
      for (int i = 0; i < 2; ++i) {
        int p = tid + (i << 8);                  // 0..511
        int r = p >> 4, c = (p & 15) << 2;       // r = k, c = n
        const float4 vb = *(const float4*)(W + (size_t)(k0 + r) * N + n0 + c);
        Bs[c + 0][r] = f32_to_bf16(vb.x);
        Bs[c + 1][r] = f32_to_bf16(vb.y);
        Bs[c + 2][r] = f32_to_bf16(vb.z);
        Bs[c + 3][r] = f32_to_bf16(vb.w);
      }
    } else {
#pragma unroll
      for (int i = 0; i < 2; ++i) {
        int p = tid + (i << 8);
        int r = p >> 4, c = (p & 15) << 2;
        int gk  = k0 + r;
        int gkc = gk < K ? gk : (K - 1);
#pragma unroll
        for (int j = 0; j < 4; ++j) {
          int gn  = n0 + c + j;
          int gnc = gn < N ? gn : (N - 1);
          float v = W[(size_t)gkc * N + gnc];
          Bs[c + j][r] = (gk < K && gn < N) ? f32_to_bf16(v) : (unsigned short)0;
        }
      }
    }
    __syncthreads();

    // prefetch next K tile (global_prefetch_b8 on gfx1250)
    if (k0 + 32 < K) {
      int pr = tid >> 1;
      if ((tid & 1) == 0 && m0 + pr < M)
        __builtin_prefetch(&A[(size_t)(m0 + pr) * lda + k0 + 32], 0, 0);
      if (tid < 32 && k0 + 32 + tid < K)
        __builtin_prefetch(&W[(size_t)(k0 + 32 + tid) * N + n0], 0, 0);
    }

    // ---- fragments: 4x b64 each ----
    Frag a0, a1, b0, b1;
    const int ar0 = (wm << 5) + rm, ar1 = ar0 + 16;
    const int bc0 = (wn << 5) + rm, bc1 = bc0 + 16;
    a0.q[0] = *(const uint2*)&As[ar0][kh];
    a0.q[1] = *(const uint2*)&As[ar0][kh + 4];
    a0.q[2] = *(const uint2*)&As[ar0][kh + 16];
    a0.q[3] = *(const uint2*)&As[ar0][kh + 20];
    a1.q[0] = *(const uint2*)&As[ar1][kh];
    a1.q[1] = *(const uint2*)&As[ar1][kh + 4];
    a1.q[2] = *(const uint2*)&As[ar1][kh + 16];
    a1.q[3] = *(const uint2*)&As[ar1][kh + 20];
    b0.q[0] = *(const uint2*)&Bs[bc0][kh];
    b0.q[1] = *(const uint2*)&Bs[bc0][kh + 4];
    b0.q[2] = *(const uint2*)&Bs[bc0][kh + 16];
    b0.q[3] = *(const uint2*)&Bs[bc0][kh + 20];
    b1.q[0] = *(const uint2*)&Bs[bc1][kh];
    b1.q[1] = *(const uint2*)&Bs[bc1][kh + 4];
    b1.q[2] = *(const uint2*)&Bs[bc1][kh + 16];
    b1.q[3] = *(const uint2*)&Bs[bc1][kh + 20];

    c00 = __builtin_amdgcn_wmma_f32_16x16x32_bf16(false, a0.v, false, b0.v, (short)0, c00, false, false);
    c01 = __builtin_amdgcn_wmma_f32_16x16x32_bf16(false, a0.v, false, b1.v, (short)0, c01, false, false);
    c10 = __builtin_amdgcn_wmma_f32_16x16x32_bf16(false, a1.v, false, b0.v, (short)0, c10, false, false);
    c11 = __builtin_amdgcn_wmma_f32_16x16x32_bf16(false, a1.v, false, b1.v, (short)0, c11, false, false);
    __syncthreads();
  }

  // ---- epilogue. C/D layout: VGPR v -> row = base + v (+8 if lane>=16), col = lane%16
  const int rb0 = m0 + (wm << 5) + ((lane >> 4) << 3);
  const int rb1 = rb0 + 16;
  const int cb0 = n0 + (wn << 5) + (lane & 15);
  const int cb1 = cb0 + 16;

  auto store_tile = [&](const floatx8& acc, int rb, int cb) {
    if (cb >= N) return;
#pragma unroll
    for (int v = 0; v < 8; ++v) {
      int r = rb + v;
      if (r >= M) continue;
      float val = acc[v];
      if (bias)  val += bias[cb];
      if (accum) val += C[(size_t)r * ldc + cb];
      if (act == 1) val = fmaxf(val, 0.0f);
      C[(size_t)r * ldc + cb] = val;
    }
  };
  store_tile(c00, rb0, cb0);
  store_tile(c01, rb0, cb1);
  store_tile(c10, rb1, cb0);
  store_tile(c11, rb1, cb1);
}

// ---------------------------------------------------------------------------
// Elementwise / graph kernels
// ---------------------------------------------------------------------------
__global__ void fill_f32(float* p, float val, size_t n) {
  size_t i = (size_t)blockIdx.x * blockDim.x + threadIdx.x;
  size_t stride = (size_t)gridDim.x * blockDim.x;
  for (; i < n; i += stride) p[i] = val;
}

__global__ void count_idx(float* cnt, const int* __restrict__ idx, int n) {
  int e = blockIdx.x * blockDim.x + threadIdx.x;
  if (e < n) atomicAdd(&cnt[idx[e]], 1.0f);
}

// agg[dst[e], :] += x[src?src[e]:e, :]
__global__ void seg_scatter_add(float* agg, const float* __restrict__ x,
                                const int* __restrict__ src,
                                const int* __restrict__ dst,
                                int nE, int dshift) {
  size_t t = (size_t)blockIdx.x * blockDim.x + threadIdx.x;
  if (t >= ((size_t)nE << dshift)) return;
  int e = (int)(t >> dshift);
  int d = (int)(t & ((1 << dshift) - 1));
  int s = src ? src[e] : e;
  atomicAdd(&agg[((size_t)dst[e] << dshift) + d], x[((size_t)s << dshift) + d]);
}

__global__ void seg_div(float* agg, const float* __restrict__ cnt, int n, int dshift) {
  size_t t = (size_t)blockIdx.x * blockDim.x + threadIdx.x;
  if (t >= ((size_t)n << dshift)) return;
  agg[t] /= fmaxf(cnt[t >> dshift], 1.0f);
}

__global__ void rsqrt_k(float* out, const float* __restrict__ in, int n) {
  int i = blockIdx.x * blockDim.x + threadIdx.x;
  if (i < n) out[i] = rsqrtf(in[i]);
}

__global__ void gcn_scatter(float* out, const float* __restrict__ h,
                            const int* __restrict__ src, const int* __restrict__ dst,
                            const float* __restrict__ dinv, int nE, int dshift) {
  size_t t = (size_t)blockIdx.x * blockDim.x + threadIdx.x;
  if (t >= ((size_t)nE << dshift)) return;
  int e = (int)(t >> dshift);
  int d = (int)(t & ((1 << dshift) - 1));
  int s = src[e], dn = dst[e];
  atomicAdd(&out[((size_t)dn << dshift) + d],
            h[((size_t)s << dshift) + d] * dinv[s] * dinv[dn]);
}

// out = out + h*dinv[i]^2 + bias[d]; optional relu
__global__ void gcn_finish(float* out, const float* __restrict__ h,
                           const float* __restrict__ dinv,
                           const float* __restrict__ bias, int n, int dshift, int act) {
  size_t t = (size_t)blockIdx.x * blockDim.x + threadIdx.x;
  if (t >= ((size_t)n << dshift)) return;
  int i = (int)(t >> dshift);
  int d = (int)(t & ((1 << dshift) - 1));
  float di = dinv[i];
  float v = out[t] + h[t] * di * di + bias[d];
  if (act) v = fmaxf(v, 0.0f);
  out[t] = v;
}

// float4-vectorized batch gathers (feature dims are multiples of 4)
__global__ void gather_rows4(float4* dst, int ldd4, const float4* __restrict__ src,
                             const int* __restrict__ idx, int n, int d4shift) {
  size_t t = (size_t)blockIdx.x * blockDim.x + threadIdx.x;
  if (t >= ((size_t)n << d4shift)) return;
  int b = (int)(t >> d4shift);
  int d = (int)(t & ((1 << d4shift) - 1));
  dst[(size_t)b * ldd4 + d] = src[((size_t)idx[b] << d4shift) + d];
}

__global__ void gather_struct4(float4* dst, int ldd4, const float4* __restrict__ uniq,
                               const int* __restrict__ map, const float4* __restrict__ miss,
                               int n, int d4shift) {
  size_t t = (size_t)blockIdx.x * blockDim.x + threadIdx.x;
  if (t >= ((size_t)n << d4shift)) return;
  int b = (int)(t >> d4shift);
  int d = (int)(t & ((1 << d4shift) - 1));
  int m = map[b];
  dst[(size_t)b * ldd4 + d] = (m >= 0) ? uniq[((size_t)m << d4shift) + d] : miss[d];
}

// rows with mask==0 get replaced by miss[d]  (mask: 1-byte jnp bool)
__global__ void mask_rows4(float4* dst, int ldd4, const unsigned char* __restrict__ mask,
                           const float4* __restrict__ miss, int n, int d4shift) {
  size_t t = (size_t)blockIdx.x * blockDim.x + threadIdx.x;
  if (t >= ((size_t)n << d4shift)) return;
  int b = (int)(t >> d4shift);
  int d = (int)(t & ((1 << d4shift) - 1));
  if (!mask[b]) dst[(size_t)b * ldd4 + d] = miss[d];
}

// feats[b,:] = full[rna_n_id[eli1[b]], :]
__global__ void gather_feats4(float4* feats, const float4* __restrict__ full,
                              const int* __restrict__ nid, const int* __restrict__ eli1,
                              int n, int d4shift) {
  size_t t = (size_t)blockIdx.x * blockDim.x + threadIdx.x;
  if (t >= ((size_t)n << d4shift)) return;
  int b = (int)(t >> d4shift);
  int d = (int)(t & ((1 << d4shift) - 1));
  feats[t] = full[((size_t)nid[eli1[b]] << d4shift) + d];
}

// LayerNorm(+act) in place, one wave32 per row. act: 0 none, 1 relu, 2 elu
__global__ void layernorm_act(float* __restrict__ X, const float* __restrict__ g,
                              const float* __restrict__ b, int rows, int d, int act) {
  int row = blockIdx.x * (blockDim.x >> 5) + (threadIdx.x >> 5);
  if (row >= rows) return;
  int lane = threadIdx.x & 31;
  float* x = X + (size_t)row * d;
  float s = 0.0f;
  for (int i = lane; i < d; i += 32) s += x[i];
  for (int o = 16; o > 0; o >>= 1) s += __shfl_xor(s, o, 32);
  float mu = s / d;
  float v = 0.0f;
  for (int i = lane; i < d; i += 32) { float t = x[i] - mu; v += t * t; }
  for (int o = 16; o > 0; o >>= 1) v += __shfl_xor(v, o, 32);
  float rstd = rsqrtf(v / d + 1e-5f);
  for (int i = lane; i < d; i += 32) {
    float y = (x[i] - mu) * rstd * g[i] + b[i];
    if (act == 1)      y = fmaxf(y, 0.0f);
    else if (act == 2) y = (y > 0.0f) ? y : expm1f(y);
    x[i] = y;
  }
}

// pred[b] = X[b,:].dot(w) + b0   (one wave32 per row)
__global__ void rowdot(const float* __restrict__ X, const float* __restrict__ w,
                       const float* __restrict__ b0, float* out, int rows, int d) {
  int row = blockIdx.x * (blockDim.x >> 5) + (threadIdx.x >> 5);
  if (row >= rows) return;
  int lane = threadIdx.x & 31;
  float s = 0.0f;
  for (int i = lane; i < d; i += 32) s += X[(size_t)row * d + i] * w[i];
  for (int o = 16; o > 0; o >>= 1) s += __shfl_xor(s, o, 32);
  if (lane == 0) out[row] = s + b0[0];
}

// ---------------------------------------------------------------------------
// Host launch
// ---------------------------------------------------------------------------
// d_in index map. Top-level inputs in setup_inputs() insertion order; the
// params pytree is flattened jax-style (dict keys sorted alphabetically,
// recursively). 76 param leaves + 12 data inputs = 88.
enum {
  P_CLF_W1 = 0, P_CLF_W2, P_CLF_B1, P_CLF_B2,                 // 0..3
  P_DNN_W1, P_DNN_W2, P_DNN_W3, P_DNN_B1, P_DNN_B2, P_DNN_B3, // 4..9
  P_DNN_G1, P_DNN_G2, P_DNN_N1, P_DNN_N2,                     // 10..13
  P_DRUG_W, P_DRUG_B,                                          // 14..15
  P_GCN1_W, P_GCN1_B, P_GCN2_W, P_GCN2_B,                      // 16..19
  P_HDA = 20,   // head_da: W1,W2,W3,b1,b2,b3,g1,g2,n1,n2
  P_HDS = 30,   // head_ds
  P_HRA = 40,   // head_ra
  P_HRS = 50,   // head_rs
  P_MISS_DRUG = 60, P_MISS_RNA = 61,
  P_RNA_W = 62, P_RNA_B = 63,
  P_S1DR = 64,  // sage1_dr: Wl,Wr,bl
  P_S1RD = 67, P_S2DR = 70, P_S2RD = 73,
  I_DRUG_X = 76, I_RNA_X, I_MOL_X, I_FULL_RNA,
  I_EI_DR = 80, I_EI_RD, I_ELI, I_RNA_NID, I_MOL_EI, I_MOL_BATCH,
  I_UMAP = 86, I_RMASK = 87
};
// within a head block: +0 W1, +1 W2, +2 W3, +3 b1, +4 b2, +5 b3, +6 g1, +7 g2, +8 n1, +9 n2
#define HW1 0
#define HW2 1
#define HW3 2
#define HB1 3
#define HB2 4
#define HB3 5
#define HG1 6
#define HG2 7
#define HN1 8
#define HN2 9

static inline dim3 ew_grid(size_t total) { return dim3((unsigned)((total + 255) / 256)); }

extern "C" void kernel_launch(void* const* d_in, const int* in_sizes, int n_in,
                              void* d_out, int out_size, void* d_ws, size_t ws_size,
                              hipStream_t stream) {
  (void)in_sizes; (void)n_in; (void)out_size; (void)ws_size;
  auto f  = [&](int i) { return (const float*)d_in[i]; };
  auto ii = [&](int i) { return (const int*)d_in[i]; };

  float* ws  = (float*)d_ws;
  float* out = (float*)d_out;

  auto gemm = [&](const float* A, int lda, const float* W, const float* bias,
                  float* C, int ldc, int M, int N, int K, int act, int accum) {
    dim3 g((N + 63) / 64, (M + 127) / 128);
    gemm_wmma_bf16<<<g, 256, 0, stream>>>(A, lda, W, bias, C, ldc, M, N, K, act, accum);
  };
  auto zero = [&](float* p, size_t n) {
    fill_f32<<<dim3(4096), 256, 0, stream>>>(p, 0.0f, n);
  };

  // ---------------- workspace layout (floats) ----------------
  const size_t XCAT = 0;                       // [B, 512] persistent concat buffer
  const size_t S0   = (size_t)BATCH * 512;     // phase-local scratch base

  // -------- Phase 1: input linears + 2-layer hetero SAGE --------
  const size_t HD = S0, HR = HD + (size_t)N_DRUG * HID,
               AGG_D = HR + (size_t)N_RNA * HID,
               AGG_R = AGG_D + (size_t)N_DRUG * HID,
               D1 = AGG_R + (size_t)N_RNA * HID,
               R1 = D1 + (size_t)N_DRUG * HID,
               D2 = R1 + (size_t)N_RNA * HID,
               R2 = D2 + (size_t)N_DRUG * OUTD,
               CNT_D = R2 + (size_t)N_RNA * OUTD,
               CNT_R = CNT_D + N_DRUG;

  const int* ei_dr = ii(I_EI_DR);            // row0 drug(src), row1 rna(dst)
  const int* ei_rd = ii(I_EI_RD);            // row0 rna(src),  row1 drug(dst)
  const int* eli0  = ii(I_ELI);              // drug indices
  const int* eli1  = ii(I_ELI) + BATCH;      // rna indices

  gemm(f(I_DRUG_X), D_DRUG,   f(P_DRUG_W), f(P_DRUG_B), ws + HD, HID, N_DRUG, HID, D_DRUG, 0, 0);
  gemm(f(I_RNA_X),  D_RNASIM, f(P_RNA_W),  f(P_RNA_B),  ws + HR, HID, N_RNA,  HID, D_RNASIM, 0, 0);

  zero(ws + CNT_D, N_DRUG);
  zero(ws + CNT_R, N_RNA);
  count_idx<<<ew_grid(NE), 256, 0, stream>>>(ws + CNT_D, ei_rd + NE, NE);
  count_idx<<<ew_grid(NE), 256, 0, stream>>>(ws + CNT_R, ei_dr + NE, NE);

  // d1 = relu(mean_{rna->drug}(hr) @ Wl + bl + hd @ Wr)
  zero(ws + AGG_D, (size_t)N_DRUG * HID);
  seg_scatter_add<<<ew_grid((size_t)NE * HID), 256, 0, stream>>>(ws + AGG_D, ws + HR, ei_rd, ei_rd + NE, NE, 8);
  seg_div<<<ew_grid((size_t)N_DRUG * HID), 256, 0, stream>>>(ws + AGG_D, ws + CNT_D, N_DRUG, 8);
  gemm(ws + AGG_D, HID, f(P_S1RD + 0), f(P_S1RD + 2), ws + D1, HID, N_DRUG, HID, HID, 0, 0);
  gemm(ws + HD,    HID, f(P_S1RD + 1), nullptr,       ws + D1, HID, N_DRUG, HID, HID, 1, 1);

  // r1 = relu(mean_{drug->rna}(hd) @ Wl + bl + hr @ Wr)
  zero(ws + AGG_R, (size_t)N_RNA * HID);
  seg_scatter_add<<<ew_grid((size_t)NE * HID), 256, 0, stream>>>(ws + AGG_R, ws + HD, ei_dr, ei_dr + NE, NE, 8);
  seg_div<<<ew_grid((size_t)N_RNA * HID), 256, 0, stream>>>(ws + AGG_R, ws + CNT_R, N_RNA, 8);
  gemm(ws + AGG_R, HID, f(P_S1DR + 0), f(P_S1DR + 2), ws + R1, HID, N_RNA, HID, HID, 0, 0);
  gemm(ws + HR,    HID, f(P_S1DR + 1), nullptr,       ws + R1, HID, N_RNA, HID, HID, 1, 1);

  // d2 = mean_{rna->drug}(r1) @ Wl + bl + d1 @ Wr     (128-d, no act)
  zero(ws + AGG_D, (size_t)N_DRUG * HID);
  seg_scatter_add<<<ew_grid((size_t)NE * HID), 256, 0, stream>>>(ws + AGG_D, ws + R1, ei_rd, ei_rd + NE, NE, 8);
  seg_div<<<ew_grid((size_t)N_DRUG * HID), 256, 0, stream>>>(ws + AGG_D, ws + CNT_D, N_DRUG, 8);
  gemm(ws + AGG_D, HID, f(P_S2RD + 0), f(P_S2RD + 2), ws + D2, OUTD, N_DRUG, OUTD, HID, 0, 0);
  gemm(ws + D1,    HID, f(P_S2RD + 1), nullptr,       ws + D2, OUTD, N_DRUG, OUTD, HID, 0, 1);

  // r2
  zero(ws + AGG_R, (size_t)N_RNA * HID);
  seg_scatter_add<<<ew_grid((size_t)NE * HID), 256, 0, stream>>>(ws + AGG_R, ws + D1, ei_dr, ei_dr + NE, NE, 8);
  seg_div<<<ew_grid((size_t)N_RNA * HID), 256, 0, stream>>>(ws + AGG_R, ws + CNT_R, N_RNA, 8);
  gemm(ws + AGG_R, HID, f(P_S2DR + 0), f(P_S2DR + 2), ws + R2, OUTD, N_RNA, OUTD, HID, 0, 0);
  gemm(ws + R1,    HID, f(P_S2DR + 1), nullptr,       ws + R2, OUTD, N_RNA, OUTD, HID, 0, 1);

  // gather assoc embeddings into xcat[:, 0:128] and [:, 128:256]  (float4)
  gather_rows4<<<ew_grid((size_t)BATCH * 32), 256, 0, stream>>>((float4*)(ws + XCAT + 0),   128, (const float4*)(ws + D2), eli0, BATCH, 5);
  gather_rows4<<<ew_grid((size_t)BATCH * 32), 256, 0, stream>>>((float4*)(ws + XCAT + 128), 128, (const float4*)(ws + R2), eli1, BATCH, 5);

  // -------- Phase 2: molecular GCN (reuses scratch) --------
  const size_t DEG = S0, DINV = DEG + N_ATOMS,
               BUFA = DINV + N_ATOMS,                       // 200000*256
               BUFB = BUFA + (size_t)N_ATOMS * HID,         // 200000*256
               UNIQ = BUFB + (size_t)N_ATOMS * HID,         // 8000*128
               CNTU = UNIQ + (size_t)N_UNIQ * OUTD;
  const size_t H2 = BUFA, OUT2 = BUFA + (size_t)N_ATOMS * OUTD;
  const int* mei = ii(I_MOL_EI);

  fill_f32<<<dim3(4096), 256, 0, stream>>>(ws + DEG, 1.0f, N_ATOMS);
  count_idx<<<ew_grid(E_MOL), 256, 0, stream>>>(ws + DEG, mei + E_MOL, E_MOL);
  rsqrt_k<<<ew_grid(N_ATOMS), 256, 0, stream>>>(ws + DINV, ws + DEG, N_ATOMS);

  // layer 1: h = mol_x @ W1 (K=78, lda=78 -> scalar edge path); out = scatter + self + bias; relu
  gemm(f(I_MOL_X), ATOM_DIM, f(P_GCN1_W), nullptr, ws + BUFA, HID, N_ATOMS, HID, ATOM_DIM, 0, 0);
  zero(ws + BUFB, (size_t)N_ATOMS * HID);
  gcn_scatter<<<ew_grid((size_t)E_MOL * HID), 256, 0, stream>>>(ws + BUFB, ws + BUFA, mei, mei + E_MOL, ws + DINV, E_MOL, 8);
  gcn_finish<<<ew_grid((size_t)N_ATOMS * HID), 256, 0, stream>>>(ws + BUFB, ws + BUFA, ws + DINV, f(P_GCN1_B), N_ATOMS, 8, 1);

  // layer 2 (128-d, no act); BUFA reused for h2 / out2
  gemm(ws + BUFB, HID, f(P_GCN2_W), nullptr, ws + H2, OUTD, N_ATOMS, OUTD, HID, 0, 0);
  zero(ws + OUT2, (size_t)N_ATOMS * OUTD);
  gcn_scatter<<<ew_grid((size_t)E_MOL * OUTD), 256, 0, stream>>>(ws + OUT2, ws + H2, mei, mei + E_MOL, ws + DINV, E_MOL, 7);
  gcn_finish<<<ew_grid((size_t)N_ATOMS * OUTD), 256, 0, stream>>>(ws + OUT2, ws + H2, ws + DINV, f(P_GCN2_B), N_ATOMS, 7, 0);

  // unique drug embedding = segment_mean(out2, mol_batch)
  zero(ws + UNIQ, (size_t)N_UNIQ * OUTD);
  zero(ws + CNTU, N_UNIQ);
  count_idx<<<ew_grid(N_ATOMS), 256, 0, stream>>>(ws + CNTU, ii(I_MOL_BATCH), N_ATOMS);
  seg_scatter_add<<<ew_grid((size_t)N_ATOMS * OUTD), 256, 0, stream>>>(ws + UNIQ, ws + OUT2, nullptr, ii(I_MOL_BATCH), N_ATOMS, 7);
  seg_div<<<ew_grid((size_t)N_UNIQ * OUTD), 256, 0, stream>>>(ws + UNIQ, ws + CNTU, N_UNIQ, 7);

  // drug_struct -> xcat[:, 256:384] (missing rows get miss_drug)
  gather_struct4<<<ew_grid((size_t)BATCH * 32), 256, 0, stream>>>((float4*)(ws + XCAT + 256), 128, (const float4*)(ws + UNIQ), ii(I_UMAP), (const float4*)f(P_MISS_DRUG), BATCH, 5);

  // -------- Phase 3: RNA sequence DNN (reuses scratch) --------
  const size_t FEATS = S0, DNH1 = FEATS + (size_t)BATCH * RNA_FEAT,
               DNH2 = DNH1 + (size_t)BATCH * DNN_HID;

  gather_feats4<<<ew_grid((size_t)BATCH * 64), 256, 0, stream>>>((float4*)(ws + FEATS), (const float4*)f(I_FULL_RNA), ii(I_RNA_NID), eli1, BATCH, 6);
  gemm(ws + FEATS, RNA_FEAT, f(P_DNN_W1), f(P_DNN_B1), ws + DNH1, DNN_HID, BATCH, DNN_HID, RNA_FEAT, 0, 0);
  layernorm_act<<<dim3(BATCH / 8), 256, 0, stream>>>(ws + DNH1, f(P_DNN_G1), f(P_DNN_N1), BATCH, DNN_HID, 1);
  gemm(ws + DNH1, DNN_HID, f(P_DNN_W2), f(P_DNN_B2), ws + DNH2, RNA_FEAT, BATCH, RNA_FEAT, DNN_HID, 0, 0);
  layernorm_act<<<dim3(BATCH / 8), 256, 0, stream>>>(ws + DNH2, f(P_DNN_G2), f(P_DNN_N2), BATCH, RNA_FEAT, 1);
  // rna_seq straight into xcat[:, 384:512], then mask-replace
  gemm(ws + DNH2, RNA_FEAT, f(P_DNN_W3), f(P_DNN_B3), ws + XCAT + 384, 512, BATCH, OUTD, RNA_FEAT, 0, 0);
  mask_rows4<<<ew_grid((size_t)BATCH * 32), 256, 0, stream>>>((float4*)(ws + XCAT + 384), 128, (const unsigned char*)d_in[I_RMASK], (const float4*)f(P_MISS_RNA), BATCH, 5);

  // -------- Phase 4: four projection heads --------
  const size_t T1 = S0, T2 = T1 + (size_t)BATCH * PROJ;
  struct HeadSpec { int in_off; int pbase; size_t out_off; };
  const HeadSpec heads[4] = {
    {256, P_HDS, 0},                          // proj_ds <- drug_struct
    {0,   P_HDA, (size_t)BATCH * OUTD},       // proj_da <- drug_assoc
    {384, P_HRS, (size_t)2 * BATCH * OUTD},   // proj_rs <- rna_seq
    {128, P_HRA, (size_t)3 * BATCH * OUTD},   // proj_ra <- rna_assoc
  };
  for (int h = 0; h < 4; ++h) {
    const float* xin = ws + XCAT + heads[h].in_off;
    int pb = heads[h].pbase;
    gemm(xin, 512, f(pb + HW1), f(pb + HB1), ws + T1, PROJ, BATCH, PROJ, OUTD, 0, 0);
    layernorm_act<<<dim3(BATCH / 8), 256, 0, stream>>>(ws + T1, f(pb + HG1), f(pb + HN1), BATCH, PROJ, 2);
    gemm(ws + T1, PROJ, f(pb + HW2), f(pb + HB2), ws + T2, PROJ, BATCH, PROJ, PROJ, 0, 0);
    layernorm_act<<<dim3(BATCH / 8), 256, 0, stream>>>(ws + T2, f(pb + HG2), f(pb + HN2), BATCH, PROJ, 2);
    gemm(ws + T2, PROJ, f(pb + HW3), f(pb + HB3), out + heads[h].out_off, OUTD, BATCH, OUTD, PROJ, 0, 0);
  }

  // -------- Phase 5: classifier on xcat --------
  const size_t HIDC = S0;
  gemm(ws + XCAT, 512, f(P_CLF_W1), f(P_CLF_B1), ws + HIDC, HID, BATCH, HID, 512, 1, 0);
  rowdot<<<dim3(BATCH / 8), 256, 0, stream>>>(ws + HIDC, f(P_CLF_W2), f(P_CLF_B2), out + (size_t)4 * BATCH * OUTD, BATCH, HID);
}
